// BYOLActiveSensor_64407329571569
// MI455X (gfx1250) — matrix operand order repaired and checked by reading, not compiled, not run on hardware
//
#include <hip/hip_runtime.h>
#include <math.h>

// ---------------------------------------------------------------------------
// PPO loss pipeline for MI455X (gfx1250, wave32, WMMA).
// GEMMs: v_wmma_f32_16x16x32_bf16, double-buffered LDS tiles, A tile staged
// via global_load_async_to_lds_b128 (ASYNCcnt), B tile transposed via VGPRs.
// ---------------------------------------------------------------------------

#define GAMMA_C 0.99f
#define LAM_C   0.95f
#define CLIP_C  0.15f
#define VF_C    0.5f
#define STD_C   0.05f
#define LOG_STD_C  (-2.9957322736f)   // ln(0.05)
#define LOG_2PI_C  (1.8378770664f)    // ln(2*pi)

#define NROWS 33280                   // 512 * 65
#define BROWS 512
#define TSTEPS 64

typedef __bf16 bf16;
typedef __attribute__((ext_vector_type(16))) __bf16 v16bf;
typedef __attribute__((ext_vector_type(8)))  __bf16 v8bf;
typedef __attribute__((ext_vector_type(8)))  float  v8f;

union V8  { v8bf v; bf16 e[8]; };
union V16 { v16bf v; v8bf h[2]; };

__device__ __forceinline__ bf16 f2bf(float x) {
  union { float f; unsigned u; } c; c.f = x;
  unsigned r = c.u + 0x7fffu + ((c.u >> 16) & 1u);   // round-to-nearest-even
  union { unsigned short s; bf16 b; } o;
  o.s = (unsigned short)(r >> 16);
  return o.b;
}

__device__ __forceinline__ bf16 bf_zero() {
  union { unsigned short s; bf16 b; } z; z.s = 0; return z.b;
}

// LDS byte offset of a generic pointer to __shared__ memory: flat LDS
// addressing uses addr[31:0] as the wave-relative logical LDS address.
__device__ __forceinline__ unsigned lds_off_of(const void* p) {
  return (unsigned)(size_t)p;
}

// Async copy 16B global -> LDS (per-lane addresses), tracked by ASYNCcnt.
__device__ __forceinline__ void async_copy_b128(unsigned lds_byte_off,
                                                const void* gaddr) {
  asm volatile("global_load_async_to_lds_b128 %0, %1, off"
               :: "v"(lds_byte_off), "v"(gaddr)
               : "memory");
}

__device__ __forceinline__ void wait_async0() {
  asm volatile("s_wait_asynccnt 0" ::: "memory");
}

// ---------------------------------------------------------------------------
// fp32 -> bf16, 8 elements/thread (n must be a multiple of 8)
// ---------------------------------------------------------------------------
__global__ void f32_to_bf16x8_kernel(const float* __restrict__ src,
                                     bf16* __restrict__ dst, int n8) {
  int i = blockIdx.x * blockDim.x + threadIdx.x;
  if (i >= n8) return;
  const float4* s4 = (const float4*)src + (size_t)i * 2;
  float4 a = s4[0], b = s4[1];
  V8 p;
  p.e[0]=f2bf(a.x); p.e[1]=f2bf(a.y); p.e[2]=f2bf(a.z); p.e[3]=f2bf(a.w);
  p.e[4]=f2bf(b.x); p.e[5]=f2bf(b.y); p.e[6]=f2bf(b.z); p.e[7]=f2bf(b.w);
  *(v8bf*)(dst + (size_t)i * 8) = p.v;
}

// ---------------------------------------------------------------------------
// WMMA GEMM: Out[M,N] = act(A[M,K]_bf16 @ W[K,N]_bf16 + bias)
//   block tile 128x64, BK=32, 256 threads = 8 waves, wave tile 32x32,
//   double-buffered LDS, async A staging.
// ---------------------------------------------------------------------------
#define BM 128
#define BN 64
#define BK 32
#define LDSS 40   // padded LDS row stride (bf16); 80B rows break bank cycles

__global__ __launch_bounds__(256)
void wmma_gemm_kernel(const bf16* __restrict__ Ap,
                      const bf16* __restrict__ W, const float* __restrict__ bias,
                      void* __restrict__ Out, int out_is_f32,
                      int M, int N, int K, int act)
{
  __shared__ __align__(16) bf16 As[2][BM * LDSS];   // [row][k]  10240 B each
  __shared__ __align__(16) bf16 Bs[2][BN * LDSS];   // [col][k]   5120 B each

  const int tid  = threadIdx.x;
  const int m0   = blockIdx.x * BM;
  const int n0   = blockIdx.y * BN;
  const int wave = tid >> 5;
  const int lane = tid & 31;
  const int lm   = lane & 15;
  const int hi   = lane >> 4;            // 0 | 1
  const int wm   = (wave & 3) * 32;      // wave row offset
  const int wn   = (wave >> 2) * 32;     // wave col offset

  // A staging: 2 threads/row, 16 bf16 (32B) each -> two async b128 copies
  const int arow = tid >> 1;             // 0..127
  const int acol = (tid & 1) * 16;       // 0 | 16
  const unsigned aDst[2] = {
    lds_off_of(&As[0][arow * LDSS + acol]),
    lds_off_of(&As[1][arow * LDSS + acol])
  };
  const bf16* aSrcRow = Ap + (size_t)(m0 + arow) * K + acol;

  // B staging: thread covers (k = tid&31, n = 8*(tid>>5)..+7), transposed store
  const int bk  = tid & 31;
  const int bn8 = (tid >> 5) * 8;

  v8f acc[2][2] = {};

  // ---- prologue: stage tile 0 into buffer 0 ----
  {
    async_copy_b128(aDst[0],      aSrcRow);
    async_copy_b128(aDst[0] + 16, aSrcRow + 8);
    const bf16* Wp = W + (size_t)bk * N + (size_t)(n0 + bn8);
    if (n0 + bn8 + 8 <= N) {
      V8 w8; w8.v = *(const v8bf*)Wp;
      #pragma unroll
      for (int j = 0; j < 8; ++j) Bs[0][(bn8 + j) * LDSS + bk] = w8.e[j];
    } else {
      #pragma unroll
      for (int j = 0; j < 8; ++j)
        Bs[0][(bn8 + j) * LDSS + bk] = (n0 + bn8 + j < N) ? Wp[j] : bf_zero();
    }
  }

  int buf = 0;
  for (int k0 = 0; k0 < K; k0 += BK, buf ^= 1) {
    wait_async0();         // own async copies into `buf` complete
    __syncthreads();       // all waves' staging into `buf` visible

    // ---- stage next tile into buf^1 (overlaps with compute below) ----
    const int kn = k0 + BK;
    if (kn < K) {
      const bf16* A = aSrcRow + kn;
      async_copy_b128(aDst[buf ^ 1],      A);
      async_copy_b128(aDst[buf ^ 1] + 16, A + 8);
      const bf16* Wp = W + (size_t)(kn + bk) * N + (size_t)(n0 + bn8);
      bf16* Bn = Bs[buf ^ 1];
      if (n0 + bn8 + 8 <= N) {
        V8 w8; w8.v = *(const v8bf*)Wp;
        #pragma unroll
        for (int j = 0; j < 8; ++j) Bn[(bn8 + j) * LDSS + bk] = w8.e[j];
      } else {
        #pragma unroll
        for (int j = 0; j < 8; ++j)
          Bn[(bn8 + j) * LDSS + bk] = (n0 + bn8 + j < N) ? Wp[j] : bf_zero();
      }
    }

    // ---- 2x2 WMMA tiles per wave on `buf` ----
    const bf16* Asb = As[buf];
    const bf16* Bsb = Bs[buf];
    #pragma unroll
    for (int tm = 0; tm < 2; ++tm) {
      // A frag (16x32): lanes 0-15 row=lm K{0..7,16..23}; lanes 16-31 K{8..15,24..31}
      const int ar  = wm + tm * 16 + lm;
      const int akb = hi * 8;
      V16 a;
      a.h[0] = *(const v8bf*)&Asb[ar * LDSS + akb];
      a.h[1] = *(const v8bf*)&Asb[ar * LDSS + akb + 16];
      #pragma unroll
      for (int tn = 0; tn < 2; ++tn) {
        // B frag (32x16): lane col=lm, lanes 0-15 K{0..7,16..23}; lanes 16-31 K{8..15,24..31}
        const int bc  = wn + tn * 16 + lm;
        const int bkb = hi * 8;
        V16 b;
        b.h[0] = *(const v8bf*)&Bsb[bc * LDSS + bkb];
        b.h[1] = *(const v8bf*)&Bsb[bc * LDSS + bkb + 16];
        acc[tm][tn] = __builtin_amdgcn_wmma_f32_16x16x32_bf16(
            false, a.v, false, b.v, (short)0, acc[tm][tn], false, false);
      }
    }
    // single barrier per iteration: next iteration's top barrier protects
    // overwriting this buffer (every wave has finished its fragment reads
    // before it can arrive there).
  }

  // ---- epilogue: bias + activation, store bf16 or fp32 ----
  #pragma unroll
  for (int tm = 0; tm < 2; ++tm) {
    #pragma unroll
    for (int tn = 0; tn < 2; ++tn) {
      const int col = n0 + wn + tn * 16 + lm;
      if (col < N) {
        const float bv = bias[col];
        const int rbase = m0 + wm + tm * 16 + hi * 8;   // C: VGPR r -> M = r + 8*hi
        #pragma unroll
        for (int r = 0; r < 8; ++r) {
          float x = acc[tm][tn][r] + bv;
          if (act == 1)      x = fmaxf(x, 0.0f);
          else if (act == 2) x = tanhf(x);
          const size_t idx = (size_t)(rbase + r) * (size_t)N + (size_t)col;
          if (out_is_f32) ((float*)Out)[idx] = x;
          else            ((bf16*)Out)[idx]  = f2bf(x);
        }
      }
    }
  }
}

// ---------------------------------------------------------------------------
// Gaussian log-prob of clipped action, summed over A=16
// ---------------------------------------------------------------------------
__global__ void logp_kernel(const float* __restrict__ mu,
                            const float* __restrict__ eps,
                            float* __restrict__ logp, int n) {
  int i = blockIdx.x * blockDim.x + threadIdx.x;
  if (i >= n) return;
  const float* m = mu  + (size_t)i * 16;
  const float* e = eps + (size_t)i * 16;
  float s = 0.0f;
  #pragma unroll
  for (int a = 0; a < 16; ++a) {
    float mm  = m[a];
    float act = fminf(fmaxf(mm + STD_C * e[a], -1.0f), 1.0f);
    float z   = (act - mm) / STD_C;
    s += -0.5f * z * z - LOG_STD_C - 0.5f * LOG_2PI_C;
  }
  logp[i] = s;
}

// ---------------------------------------------------------------------------
// sigma_r = sqrt(max(E[r^2]-E[r]^2, 0) + 1e-8)   (single block)
// ---------------------------------------------------------------------------
__global__ __launch_bounds__(256)
void reward_stats_kernel(const float* __restrict__ r, int n,
                         float* __restrict__ sigma_out) {
  __shared__ float s1[256], s2[256];
  const int tid = threadIdx.x;
  float a = 0.0f, b = 0.0f;
  for (int i = tid; i < n; i += 256) { float x = r[i]; a += x; b += x * x; }
  s1[tid] = a; s2[tid] = b;
  __syncthreads();
  for (int off = 128; off > 0; off >>= 1) {
    if (tid < off) { s1[tid] += s1[tid + off]; s2[tid] += s2[tid + off]; }
    __syncthreads();
  }
  if (tid == 0) {
    float m  = s1[0] / (float)n;
    float m2 = s2[0] / (float)n;
    sigma_out[0] = sqrtf(fmaxf(m2 - m * m, 0.0f) + 1e-8f);
  }
}

// ---------------------------------------------------------------------------
// GAE scan + clipped value loss + PPO surrogate, full reduction to scalar.
// One 512-thread block; thread b owns batch row b.
// ---------------------------------------------------------------------------
__global__ __launch_bounds__(512)
void final_loss_kernel(const float* __restrict__ log_probs,
                       const float* __restrict__ rewards,
                       const float* __restrict__ values,
                       const float* __restrict__ relogp,
                       const float* __restrict__ revalues,
                       const float* __restrict__ sigma_p,
                       float* __restrict__ advbuf,
                       float* __restrict__ out) {
  __shared__ float sA[512], sV[512];
  const int b = threadIdx.x;
  const float inv_sigma = 1.0f / sigma_p[0];

  const float* rw  = rewards + (size_t)b * (TSTEPS + 1);
  const float* vl  = values  + (size_t)b * (TSTEPS + 1);
  float*       adv = advbuf  + (size_t)b * (TSTEPS + 1);

  // reverse GAE scan; accumulate mean/var stats over adv[1..T]
  float gae = rw[TSTEPS] * inv_sigma - vl[TSTEPS];
  adv[TSTEPS] = gae;
  float sum = gae, sumsq = gae * gae;
  for (int t = TSTEPS - 1; t >= 0; --t) {
    gae = rw[t] * inv_sigma + GAMMA_C * vl[t + 1] - vl[t] + GAMMA_C * LAM_C * gae;
    adv[t] = gae;
    if (t >= 1) { sum += gae; sumsq += gae * gae; }
  }
  const float mean = sum / (float)TSTEPS;
  const float var  = (sumsq - (float)TSTEPS * mean * mean) / (float)(TSTEPS - 1);
  const float sd   = sqrtf(fmaxf(var, 0.0f));

  // clipped value loss over t = 0..T
  float vsum = 0.0f;
  for (int t = 0; t <= TSTEPS; ++t) {
    float rv  = revalues[(size_t)b * (TSTEPS + 1) + t];
    float v0  = vl[t];
    float tgt = adv[t] + v0;
    float d1  = rv - tgt;
    float dv  = fminf(fmaxf(rv - v0, -CLIP_C), CLIP_C);
    float d2  = (v0 + dv) - tgt;
    vsum += fmaxf(d1 * d1, d2 * d2);
  }

  // PPO surrogate over t = 0..T-1
  float asum = 0.0f;
  for (int t = 0; t < TSTEPS; ++t) {
    float ratio = expf(relogp[(size_t)b * (TSTEPS + 1) + t] -
                       log_probs[(size_t)b * (TSTEPS + 1) + t + 1]);
    float g  = (adv[t + 1] - mean) / (sd + 1e-8f);
    float rc = fminf(fmaxf(ratio, 1.0f - CLIP_C), 1.0f + CLIP_C);
    asum += fminf(ratio * g, rc * g);
  }

  sA[b] = asum; sV[b] = vsum;
  __syncthreads();
  for (int off = 256; off > 0; off >>= 1) {
    if (b < off) { sA[b] += sA[b + off]; sV[b] += sV[b + off]; }
    __syncthreads();
  }
  if (b == 0) {
    float actor_loss = -sA[0] / (float)(BROWS * TSTEPS);
    float value_loss = 0.5f * sV[0] / (float)(BROWS * (TSTEPS + 1));
    out[0] = actor_loss + VF_C * value_loss;
  }
}

// ---------------------------------------------------------------------------
// Host launcher
// ---------------------------------------------------------------------------
extern "C" void kernel_launch(void* const* d_in, const int* in_sizes, int n_in,
                              void* d_out, int out_size, void* d_ws, size_t ws_size,
                              hipStream_t stream) {
  (void)in_sizes; (void)n_in; (void)out_size; (void)ws_size;
  char* wsb = (char*)d_ws;
  const int NR = NROWS;

  size_t off = 0;
  auto take = [&](size_t bytes) {
    size_t cur = off;
    off += (bytes + 255) & ~(size_t)255;
    return cur;
  };
  // bf16 states + weights
  size_t o_sbf  = take((size_t)NR * 1024 * 2);
  size_t o_wAe0 = take((size_t)1024 * 256 * 2);
  size_t o_wAe1 = take((size_t)256 * 256 * 2);
  size_t o_wAe2 = take((size_t)256 * 512 * 2);
  size_t o_wCe0 = take((size_t)1024 * 256 * 2);
  size_t o_wCe1 = take((size_t)256 * 256 * 2);
  size_t o_wCe2 = take((size_t)256 * 512 * 2);
  size_t o_wAm0 = take((size_t)512 * 64 * 2);
  size_t o_wAm1 = take((size_t)64 * 16 * 2);
  size_t o_wCr0 = take((size_t)512 * 64 * 2);
  size_t o_wCr1 = take((size_t)64 * 1 * 2);
  // activations (actor/critic reuse the same buffers sequentially)
  size_t o_h0  = take((size_t)NR * 256 * 2);
  size_t o_h1  = take((size_t)NR * 256 * 2);
  size_t o_z   = take((size_t)NR * 512 * 2);
  size_t o_hh  = take((size_t)NR * 64 * 2);
  size_t o_mu  = take((size_t)NR * 16 * 4);
  size_t o_lp  = take((size_t)NR * 4);
  size_t o_v   = take((size_t)NR * 4);
  size_t o_adv = take((size_t)NR * 4);
  size_t o_sig = take(64);

  auto conv = [&](const void* src, size_t dst_off, size_t n) {
    int n8 = (int)(n / 8);
    f32_to_bf16x8_kernel<<<dim3((n8 + 255) / 256), dim3(256), 0, stream>>>(
        (const float*)src, (bf16*)(wsb + dst_off), n8);
  };
  conv(d_in[0],  o_sbf,  (size_t)NR * 1024);
  conv(d_in[5],  o_wAe0, 1024 * 256);
  conv(d_in[7],  o_wAe1, 256 * 256);
  conv(d_in[9],  o_wAe2, 256 * 512);
  conv(d_in[11], o_wCe0, 1024 * 256);
  conv(d_in[13], o_wCe1, 256 * 256);
  conv(d_in[15], o_wCe2, 256 * 512);
  conv(d_in[17], o_wAm0, 512 * 64);
  conv(d_in[19], o_wAm1, 64 * 16);
  conv(d_in[21], o_wCr0, 512 * 64);
  conv(d_in[23], o_wCr1, 64 * 1);

  auto gemm = [&](size_t a_off, size_t w_off, int bias_idx,
                  void* Outp, int o32, int N, int K, int act) {
    dim3 grid(NR / BM, (N + BN - 1) / BN);
    wmma_gemm_kernel<<<grid, dim3(256), 0, stream>>>(
        (const bf16*)(wsb + a_off), (const bf16*)(wsb + w_off),
        (const float*)d_in[bias_idx], Outp, o32, NR, N, K, act);
  };

  // ---- actor path ----
  gemm(o_sbf, o_wAe0, 6,  wsb + o_h0, 0, 256, 1024, 1);  // relu
  gemm(o_h0,  o_wAe1, 8,  wsb + o_h1, 0, 256, 256,  1);  // relu
  gemm(o_h1,  o_wAe2, 10, wsb + o_z,  0, 512, 256,  2);  // tanh -> z_a
  gemm(o_z,   o_wAm0, 18, wsb + o_hh, 0, 64,  512,  1);  // relu
  gemm(o_hh,  o_wAm1, 20, wsb + o_mu, 1, 16,  64,   2);  // tanh -> mu (f32)
  logp_kernel<<<dim3((NR + 255) / 256), dim3(256), 0, stream>>>(
      (const float*)(wsb + o_mu), (const float*)d_in[4],
      (float*)(wsb + o_lp), NR);

  // ---- critic path (reuses activation buffers) ----
  gemm(o_sbf, o_wCe0, 12, wsb + o_h0, 0, 256, 1024, 1);
  gemm(o_h0,  o_wCe1, 14, wsb + o_h1, 0, 256, 256,  1);
  gemm(o_h1,  o_wCe2, 16, wsb + o_z,  0, 512, 256,  2);
  gemm(o_z,   o_wCr0, 22, wsb + o_hh, 0, 64,  512,  1);
  gemm(o_hh,  o_wCr1, 24, wsb + o_v,  1, 1,   64,   0);  // -> v (f32)

  // ---- reward normalization + GAE + losses ----
  reward_stats_kernel<<<dim3(1), dim3(256), 0, stream>>>(
      (const float*)d_in[2], NR, (float*)(wsb + o_sig));
  final_loss_kernel<<<dim3(1), dim3(512), 0, stream>>>(
      (const float*)d_in[1], (const float*)d_in[2], (const float*)d_in[3],
      (const float*)(wsb + o_lp), (const float*)(wsb + o_v),
      (const float*)(wsb + o_sig), (float*)(wsb + o_adv), (float*)d_out);
}
